// PromptDurationMemoryEncoder_79491254714841
// MI455X (gfx1250) — matrix operand order repaired and checked by reading, not compiled, not run on hardware
//
#include <hip/hip_runtime.h>
#include <hip/hip_bf16.h>
#include <math.h>

// ---------------- problem constants ----------------
#define BB 32
#define TT 2048
#define DD 512
#define VV 1024
#define KK 3
#define BT (BB*TT)           // 65536

// ---------------- d_out flat offsets (floats) ----------------
#define O_GR     0            // [B]
#define O_SUM    32           // [B,D]
#define O_RMEAN  16416        // [B]
#define O_RVAR   16448        // [B]
#define O_COV    16480        // [B]
#define O_MASK   16512        // [B,T]
#define O_LOGDUR 82048        // [B,T]
#define O_RESID  147584       // [B,T]
#define O_ATTN   213120       // [B,T]
#define O_FIT    278656       // [B,T]
#define O_CNORM  344192       // [B]

// ---------------- types ----------------
typedef __attribute__((ext_vector_type(16))) __bf16 bf16x16;
typedef __attribute__((ext_vector_type(8)))  __bf16 bf16x8;
typedef __attribute__((ext_vector_type(8)))  float  f32x8;
typedef __attribute__((ext_vector_type(4)))  unsigned int u32x4;
typedef __attribute__((ext_vector_type(8)))  int    i32x8;
typedef __attribute__((ext_vector_type(4)))  int    i32x4;

#if defined(__has_builtin)
# if __has_builtin(__builtin_amdgcn_tensor_load_to_lds) && __has_builtin(__builtin_amdgcn_s_wait_tensorcnt)
#  define HAVE_TDM 1
# endif
#endif

__device__ __forceinline__ unsigned short f2bf(float f) {
    unsigned u = __float_as_uint(f);
    u += 0x7FFFu + ((u >> 16) & 1u);          // round-to-nearest-even
    return (unsigned short)(u >> 16);
}
__device__ __forceinline__ float bf2f(unsigned short s) {
    return __uint_as_float(((unsigned)s) << 16);
}
__device__ __forceinline__ float gelu_erf(float x) {
    return 0.5f * x * (1.0f + erff(x * 0.70710678118654752f));
}
__device__ __forceinline__ f32x8 wmma_bf16(bf16x16 a, bf16x16 b, f32x8 c) {
    // emits v_wmma_f32_16x16x32_bf16
    return __builtin_amdgcn_wmma_f32_16x16x32_bf16(false, a, false, b, (short)0, c, false, false);
}

// =====================================================================
// Phase 0: per-batch masked median (bitonic sort), residual stats,
// and the 5 cheap per-timestep outputs.
// One block per batch row, 256 threads.
// =====================================================================
__global__ __launch_bounds__(256) void stats_kernel(
    const float* __restrict__ dur, const float* __restrict__ maskin,
    float* __restrict__ out, float* __restrict__ support_ws)
{
    __shared__ float v[TT];
    __shared__ int   scnt;
    __shared__ float sgr, ssum, svsum;
    const int b = blockIdx.x, tid = threadIdx.x;
    if (tid == 0) { scnt = 0; ssum = 0.0f; svsum = 0.0f; }
    __syncthreads();

    int localcnt = 0;
    for (int t = tid; t < TT; t += 256) {
        float m  = fminf(fmaxf(maskin[b*TT + t], 0.0f), 1.0f);
        float ld = logf(fmaxf(dur[b*TT + t], 1e-4f)) * m;
        out[O_MASK   + b*TT + t] = m;
        out[O_LOGDUR + b*TT + t] = ld;
        v[t] = (m > 0.5f) ? ld : __builtin_inff();
        if (m > 0.5f) localcnt++;
    }
    atomicAdd(&scnt, localcnt);
    __syncthreads();

    // bitonic sort of 2048 floats, ascending
    for (int k = 2; k <= TT; k <<= 1) {
        for (int j = k >> 1; j > 0; j >>= 1) {
            for (int i = tid; i < TT; i += 256) {
                int ixj = i ^ j;
                if (ixj > i) {
                    float a = v[i], c = v[ixj];
                    bool up = ((i & k) == 0);
                    if ((a > c) == up) { v[i] = c; v[ixj] = a; }
                }
            }
            __syncthreads();
        }
    }
    if (tid == 0) {
        int n = scnt;
        float med = (n > 0) ? v[(n - 1) >> 1] : 0.0f;
        sgr = med;
        out[O_GR + b] = med;
        support_ws[b] = (float)n;
    }
    __syncthreads();
    const float gr = sgr;

    float lsum = 0.0f;
    for (int t = tid; t < TT; t += 256) {
        float m  = out[O_MASK   + b*TT + t];
        float ld = out[O_LOGDUR + b*TT + t];
        float r  = (ld - gr) * m;
        out[O_RESID + b*TT + t] = r;
        lsum += r;
    }
    atomicAdd(&ssum, lsum);
    __syncthreads();

    const float support = (float)scnt;
    const float denom   = fmaxf(support, 1.0f);
    const float rm      = ssum / denom;

    float lv = 0.0f;
    for (int t = tid; t < TT; t += 256) {
        float m = out[O_MASK  + b*TT + t];
        float r = out[O_RESID + b*TT + t];
        float d = r - rm;
        lv += d * d * m;
        out[O_ATTN + b*TT + t] = m / denom;
        out[O_FIT  + b*TT + t] = rm * m;
    }
    atomicAdd(&svsum, lv);
    __syncthreads();
    if (tid == 0) {
        out[O_RMEAN + b] = rm;
        out[O_RVAR  + b] = fmaxf(svsum / denom, 1e-4f);
        out[O_COV   + b] = fmaxf(support / (float)TT, 0.05f);
    }
}

// =====================================================================
// Weight repack: w[o][i][tap] fp32 -> wT[tap][i][o] bf16
// =====================================================================
__global__ __launch_bounds__(256) void wtrans_kernel(
    const float* __restrict__ w, unsigned short* __restrict__ wT)
{
    int idx = blockIdx.x * 256 + threadIdx.x;
    if (idx >= KK * DD * DD) return;
    int tap = idx / (DD * DD);
    int rem = idx % (DD * DD);
    int i = rem / DD;
    int o = rem % DD;
    wT[idx] = f2bf(w[(o * DD + i) * KK + tap]);
}

// =====================================================================
// Embedding + aux injection -> bf16 activation buffer
// One block per (b,t) row.
// =====================================================================
__global__ __launch_bounds__(256) void embed_kernel(
    const int* __restrict__ ids, const float* __restrict__ resid,
    const float* __restrict__ emb, const float* __restrict__ aux_w,
    const float* __restrict__ aux_b, unsigned short* __restrict__ outbuf)
{
    const int row = blockIdx.x;
    const int id  = ids[row];
    const float r = resid[row];
    for (int d = threadIdx.x; d < DD; d += 256) {
        float vv = emb[(size_t)id * DD + d] + r * aux_w[d] + aux_b[d];
        outbuf[(size_t)row * DD + d] = f2bf(vv);
    }
}

// =====================================================================
// Causal conv (K=3) as GEMM  y = sum_tap X(shift tap-2) @ W_tap^T
// WMMA bf16, fused bias + exact-erf GELU + bf16 store.
// Weight panels staged via Tensor Data Mover (tensor_load_to_lds),
// X panel via vector loads (handles causal zero-fill at batch start).
// Grid: (BT/128, D/128), 256 threads = 8 wave32.
// =====================================================================
__global__ __launch_bounds__(256) void conv_wmma_kernel(
    const unsigned short* __restrict__ x,    // [BT, D] bf16
    const unsigned short* __restrict__ wT,   // [3, D, D] bf16  (tap, i, o)
    const float* __restrict__ bias,          // [D]
    unsigned short* __restrict__ y)          // [BT, D] bf16
{
    const int tid  = threadIdx.x;
    const int lane = tid & 31;
    const int wave = tid >> 5;
    const int wm   = wave & 1;     // M half (64 rows)
    const int wn   = wave >> 1;    // N quarter (32 cols)
    const int m0   = blockIdx.x * 128;
    const int n0   = blockIdx.y * 128;
    const int t0   = m0 & (TT - 1);
    const int rowOrg = m0 - t0;    // batch*T

    __shared__ __align__(16) unsigned short sX[130 * 40];      // 130 rows, stride 40 bf16
    __shared__ __align__(16) unsigned short sW[3 * 32 * 128];  // [tap][k][n]

    f32x8 acc[4][2];
    for (int i = 0; i < 4; ++i)
        for (int j = 0; j < 2; ++j)
            for (int e = 0; e < 8; ++e) acc[i][j][e] = 0.0f;

    const int mrow = lane & 15;
    const int lh   = lane >> 4;

#ifdef HAVE_TDM
    // ---- Tensor DMA descriptor (D#) for the weight panel (ISA ch.8) ----
    // 3D tile: x = 128 outputs (contig), y = 32 input-channels (stride 512),
    //          z = 3 taps (stride 512*512); elements are 2 bytes.
    // LDS destination is contiguous [tap][k][n] = 24576 B.
    const unsigned ldsW = (unsigned)(unsigned long long)(uintptr_t)(void*)sW;
    u32x4 g0; i32x8 g1; i32x4 g2, g3; i32x8 gz;
    g0[0] = 1u;                 // count=1, user descriptor
    g0[1] = ldsW;               // lds_addr
    g1[0] = 0x00010000;         // data_size=2B, no multicast/iterate/pad
    g1[1] = (int)(512u << 16);  // tensor_dim0[15:0] = 512
    g1[2] = (int)(512u << 16);  // tensor_dim1[15:0] = 512
    g1[3] = (int)(128u << 16);  // tile_dim0 = 128
    g1[4] = (int)((3u << 16) | 32u);  // tile_dim2=3, tile_dim1=32
    g1[5] = 512;                // tensor_dim0_stride[31:0]
    g1[6] = 0;                  // stride0 hi / stride1 lo
    g1[7] = 4;                  // tensor_dim1_stride[47:16] = 262144>>16
    g2[0] = 3; g2[1] = 0; g2[2] = 0; g2[3] = 0;   // tensor_dim2=3, dims 3/4 unused
    g3 = g2;
    for (int e = 0; e < 8; ++e) gz[e] = 0;
#endif

    for (int kc = 0; kc < DD; kc += 32) {
        __syncthreads();
#ifdef HAVE_TDM
        // ---- issue TDM for the W panel from wave 0 (overlaps X staging) ----
        if (wave == 0) {
            unsigned long long ga = (unsigned long long)(uintptr_t)wT
                                  + 2ull * ((unsigned long long)kc * 512ull
                                            + (unsigned long long)n0);
            g0[2] = (unsigned)ga;
            g0[3] = (unsigned)((ga >> 32) & 0x01FFFFFFull) | (2u << 30); // type=2
            __builtin_amdgcn_tensor_load_to_lds(g0, g1, g2, g3, gz, 0);
        }
#else
        // ---- fallback: stage W panel [3][32][128] with vector loads ----
        for (int idx = tid; idx < 3 * 32 * 16; idx += 256) {
            int rr = idx >> 4, c = idx & 15;          // rr = tap*32 + k
            int tap = rr >> 5, k = rr & 31;
            *(uint4*)(sW + rr * 128 + c * 8) =
                *(const uint4*)(wT + ((size_t)tap * DD * DD + (size_t)(kc + k) * DD + n0 + c * 8));
        }
#endif
        // ---- stage X panel: rows [m0-2, m0+128) x cols [kc, kc+32) ----
        for (int idx = tid; idx < 130 * 4; idx += 256) {
            int r = idx >> 2, c4 = idx & 3;
            int t = t0 - 2 + r;
            uint4 val = make_uint4(0u, 0u, 0u, 0u);
            if (t >= 0)
                val = *(const uint4*)(x + ((size_t)(rowOrg + t) * DD + kc + c4 * 8));
            *(uint4*)(sX + r * 40 + c4 * 8) = val;
        }
        // prefetch next K-chunk of X (global_prefetch_b8)
        if (kc + 32 < DD && tid < 128) {
            int tp = t0 - 2 + tid;
            if (tp >= 0)
                __builtin_prefetch((const void*)(x + ((size_t)(rowOrg + tp) * DD + kc + 32)), 0, 1);
        }
#ifdef HAVE_TDM
        if (wave == 0) __builtin_amdgcn_s_wait_tensorcnt(0);
#endif
        __syncthreads();

        for (int tap = 0; tap < 3; ++tap) {
            bf16x16 afrag[4];
            for (int i = 0; i < 4; ++i) {
                int r = wm * 64 + i * 16 + mrow + tap;          // shifted row in panel
                bf16x8 lo = *(const bf16x8*)(sX + r * 40 + lh * 8);
                bf16x8 hi = *(const bf16x8*)(sX + r * 40 + 16 + lh * 8);
                afrag[i] = __builtin_shufflevector(lo, hi,
                    0,1,2,3,4,5,6,7,8,9,10,11,12,13,14,15);
            }
            for (int j = 0; j < 2; ++j) {
                int nc = wn * 32 + j * 16;
                bf16x16 bfrag;
                for (int s = 0; s < 16; ++s) {
                    int kk = (s < 8) ? (lh * 8 + s) : (16 + lh * 8 + (s - 8));
                    bfrag[s] = *(const __bf16*)(sW + (tap * 32 + kk) * 128 + nc + mrow);
                }
                for (int i = 0; i < 4; ++i)
                    acc[i][j] = wmma_bf16(afrag[i], bfrag, acc[i][j]);
            }
        }
    }

    // ---- epilogue: bias + GELU + bf16 store ----
    for (int i = 0; i < 4; ++i)
        for (int j = 0; j < 2; ++j) {
            int col = n0 + wn * 32 + j * 16 + (lane & 15);
            float bv = bias[col];
            for (int e = 0; e < 8; ++e) {
                int row = m0 + wm * 64 + i * 16 + e + lh * 8;
                float g = gelu_erf(acc[i][j][e] + bv);
                y[(size_t)row * DD + col] = f2bf(g);
            }
        }
}

// =====================================================================
// LayerNorm over D + mask multiply. One wave32 per row.
// =====================================================================
__global__ __launch_bounds__(256) void ln_kernel(
    const unsigned short* __restrict__ x, const float* __restrict__ mask,
    const float* __restrict__ g, const float* __restrict__ bvec,
    unsigned short* __restrict__ y)
{
    const int tid  = threadIdx.x;
    const int lane = tid & 31;
    const int row  = blockIdx.x * 8 + (tid >> 5);

    const size_t base = (size_t)row * DD + lane * 16;
    uint4 q0 = *(const uint4*)(x + base);
    uint4 q1 = *(const uint4*)(x + base + 8);

    float vals[16];
    unsigned qs[8] = {q0.x, q0.y, q0.z, q0.w, q1.x, q1.y, q1.z, q1.w};
    for (int e = 0; e < 8; ++e) {
        vals[2*e]   = bf2f((unsigned short)(qs[e] & 0xFFFFu));
        vals[2*e+1] = bf2f((unsigned short)(qs[e] >> 16));
    }
    float sum = 0.0f, sq = 0.0f;
    for (int e = 0; e < 16; ++e) { sum += vals[e]; sq += vals[e] * vals[e]; }
    for (int off = 16; off > 0; off >>= 1) {
        sum += __shfl_xor(sum, off, 32);
        sq  += __shfl_xor(sq,  off, 32);
    }
    const float mean = sum * (1.0f / DD);
    const float var  = sq * (1.0f / DD) - mean * mean;
    const float rstd = rsqrtf(var + 1e-5f);
    const float mk   = mask[row];
    for (int e = 0; e < 16; ++e) {
        int d = lane * 16 + e;
        float yy = ((vals[e] - mean) * rstd * g[d] + bvec[d]) * mk;
        y[(size_t)row * DD + d] = f2bf(yy);
    }
}

// =====================================================================
// Masked pooling over T: mean + std per (b,d). One block per batch row.
// =====================================================================
__global__ __launch_bounds__(256) void pool_kernel(
    const unsigned short* __restrict__ ln, const float* __restrict__ support_ws,
    float* __restrict__ pooled)
{
    const int b = blockIdx.x, tid = threadIdx.x;
    const int d0 = tid, d1 = tid + 256;
    float s1a = 0.0f, s2a = 0.0f, s1b = 0.0f, s2b = 0.0f;
    for (int t = 0; t < TT; ++t) {
        const size_t base = ((size_t)b * TT + t) * DD;
        float xa = bf2f(ln[base + d0]);
        float xb = bf2f(ln[base + d1]);
        s1a += xa; s2a += xa * xa;
        s1b += xb; s2b += xb * xb;
    }
    const float support = support_ws[b];
    const float denom   = fmaxf(support, 1.0f);
    float ma = s1a / denom;
    float mb = s1b / denom;
    float sa = sqrtf((s2a - 2.0f * ma * s1a + ma * ma * support) / denom + 1e-6f);
    float sb = sqrtf((s2b - 2.0f * mb * s1b + mb * mb * support) / denom + 1e-6f);
    pooled[b * 1024 + d0]       = ma;
    pooled[b * 1024 + d1]       = mb;
    pooled[b * 1024 + DD + d0]  = sa;
    pooled[b * 1024 + DD + d1]  = sb;
}

// =====================================================================
// MLP head (tiny): y1 = gelu(h @ p1^T + b1); y2 = tanh(y1 @ p2^T + b2)
// =====================================================================
__global__ __launch_bounds__(256) void mlp1_kernel(
    const float* __restrict__ pooled, const float* __restrict__ w,
    const float* __restrict__ bvec, float* __restrict__ y1)
{
    int idx = blockIdx.x * 256 + threadIdx.x;   // 32*512
    int b = idx >> 9, o = idx & 511;
    const float* hrow = pooled + b * 1024;
    const float* wrow = w + (size_t)o * 1024;
    float s = bvec[o];
    for (int i = 0; i < 1024; ++i) s += hrow[i] * wrow[i];
    y1[idx] = gelu_erf(s);
}

__global__ __launch_bounds__(256) void mlp2_kernel(
    const float* __restrict__ y1, const float* __restrict__ w,
    const float* __restrict__ bvec, const float* __restrict__ support_ws,
    float* __restrict__ out)
{
    int idx = blockIdx.x * 256 + threadIdx.x;
    int b = idx >> 9, o = idx & 511;
    const float* hrow = y1 + b * DD;
    const float* wrow = w + (size_t)o * DD;
    float s = bvec[o];
    for (int i = 0; i < DD; ++i) s += hrow[i] * wrow[i];
    float v = tanhf(s);
    if (support_ws[b] <= 0.0f) v = 0.0f;
    out[O_SUM + idx] = v;
}

__global__ __launch_bounds__(256) void norm_kernel(float* __restrict__ out)
{
    __shared__ float red[256];
    const int b = blockIdx.x, tid = threadIdx.x;
    float s = 0.0f;
    for (int d = tid; d < DD; d += 256) {
        float v = out[O_SUM + b * DD + d];
        s += v * v;
    }
    red[tid] = s;
    __syncthreads();
    for (int st = 128; st > 0; st >>= 1) {
        if (tid < st) red[tid] += red[tid + st];
        __syncthreads();
    }
    if (tid == 0) out[O_CNORM + b] = sqrtf(red[0]);
}

// =====================================================================
extern "C" void kernel_launch(void* const* d_in, const int* in_sizes, int n_in,
                              void* d_out, int out_size, void* d_ws, size_t ws_size,
                              hipStream_t stream)
{
    const int*   unit_ids = (const int*)  d_in[0];
    const float* dur      = (const float*)d_in[1];
    const float* maskin   = (const float*)d_in[2];
    const float* emb      = (const float*)d_in[3];
    const float* aux_w    = (const float*)d_in[4];
    const float* aux_b    = (const float*)d_in[5];
    const float* c1w      = (const float*)d_in[6];
    const float* c1b      = (const float*)d_in[7];
    const float* c2w      = (const float*)d_in[8];
    const float* c2b      = (const float*)d_in[9];
    const float* ln_g     = (const float*)d_in[10];
    const float* ln_b     = (const float*)d_in[11];
    const float* p1w      = (const float*)d_in[12];
    const float* p1b      = (const float*)d_in[13];
    const float* p2w      = (const float*)d_in[14];
    const float* p2b      = (const float*)d_in[15];
    float* out = (float*)d_out;

    // workspace carve-up
    unsigned short* bufA = (unsigned short*)d_ws;               // [BT,D] bf16
    unsigned short* bufB = bufA + (size_t)BT * DD;              // [BT,D] bf16
    unsigned short* wT1  = bufB + (size_t)BT * DD;              // [3,D,D] bf16
    unsigned short* wT2  = wT1 + (size_t)KK * DD * DD;
    float* pooled  = (float*)(wT2 + (size_t)KK * DD * DD);      // [B,2D]
    float* y1      = pooled + BB * 2 * DD;                      // [B,D]
    float* support = y1 + BB * DD;                              // [B]

    stats_kernel<<<BB, 256, 0, stream>>>(dur, maskin, out, support);
    wtrans_kernel<<<(KK * DD * DD + 255) / 256, 256, 0, stream>>>(c1w, wT1);
    wtrans_kernel<<<(KK * DD * DD + 255) / 256, 256, 0, stream>>>(c2w, wT2);
    embed_kernel<<<BT, 256, 0, stream>>>(unit_ids, out + O_RESID, emb, aux_w, aux_b, bufA);

    dim3 cgrid(BT / 128, DD / 128);
    conv_wmma_kernel<<<cgrid, 256, 0, stream>>>(bufA, wT1, c1b, bufB);
    conv_wmma_kernel<<<cgrid, 256, 0, stream>>>(bufB, wT2, c2b, bufA);

    ln_kernel<<<BT / 8, 256, 0, stream>>>(bufA, out + O_MASK, ln_g, ln_b, bufB);
    pool_kernel<<<BB, 256, 0, stream>>>(bufB, support, pooled);
    mlp1_kernel<<<(BB * DD) / 256, 256, 0, stream>>>(pooled, p1w, p1b, y1);
    mlp2_kernel<<<(BB * DD) / 256, 256, 0, stream>>>(y1, p2w, p2b, support, out);
    norm_kernel<<<BB, 256, 0, stream>>>(out);
}
